// Elman_32349693673945
// MI455X (gfx1250) — matrix operand order repaired and checked by reading, not compiled, not run on hardware
//
#include <hip/hip_runtime.h>

typedef __attribute__((ext_vector_type(16))) __bf16 v16bf;
typedef __attribute__((ext_vector_type(8)))  float  v8f;

union ABFrag { v16bf v; uint4 q[2]; unsigned u[8]; };

#define B_ 64
#define T_ 512
#define E_ 512
#define H_ 512
#define O_ 512
#define BTO ((size_t)B_ * T_ * O_)   // 16777216 floats of outs, then hidden tail

__device__ __forceinline__ unsigned pack2_bf16(float lo, float hi) {
  unsigned a = __float_as_uint(lo), b = __float_as_uint(hi);
  a = (a + 0x7FFFu + ((a >> 16) & 1u)) >> 16;   // round-to-nearest-even
  b = (b + 0x7FFFu + ((b >> 16) & 1u)) >> 16;
  return (a & 0xFFFFu) | (b << 16);
}
__device__ __forceinline__ unsigned short f32_to_bf16(float x) {
  unsigned u = __float_as_uint(x);
  return (unsigned short)((u + 0x7FFFu + ((u >> 16) & 1u)) >> 16);
}

// ---------------------------------------------------------------------------
// init: zero hidden double-buffer + grid-barrier counter (re-run every call)
// ---------------------------------------------------------------------------
__global__ void elman_init(unsigned* __restrict__ hzero, unsigned* __restrict__ ctr) {
  unsigned i = blockIdx.x * blockDim.x + threadIdx.x;
  if (i == 0) *ctr = 0u;
  for (unsigned idx = i; idx < 2u * B_ * H_ / 2u; idx += blockDim.x * gridDim.x)
    hzero[idx] = 0u;   // 2 * 64*512 ushorts = 32768 u32
}

// ---------------------------------------------------------------------------
// Pass 0 (memory-bound): x [B,T,E] f32  ->  Xb [m][k] bf16, m = t*64 + b.
// One thread = 8 elements: two float4 loads, one uint4 store.
// ---------------------------------------------------------------------------
__global__ __launch_bounds__(256) void elman_cvt_x(const float* __restrict__ x,
                                                   unsigned* __restrict__ Xb) {
  unsigned tid = blockIdx.x * blockDim.x + threadIdx.x;   // 2,097,152 threads
  unsigned e  = tid * 8;                                  // flat bf16 element idx
  unsigned m  = e >> 9, k0 = e & 511;
  unsigned bb = m & 63, tt = m >> 6;
  const float* p = x + ((size_t)bb * T_ + tt) * E_ + k0;
  float4 f0 = *(const float4*)(p);
  float4 f1 = *(const float4*)(p + 4);
  uint4 o;
  o.x = pack2_bf16(f0.x, f0.y); o.y = pack2_bf16(f0.z, f0.w);
  o.z = pack2_bf16(f1.x, f1.y); o.w = pack2_bf16(f1.z, f1.w);
  ((uint4*)Xb)[tid] = o;
}

// ---------------------------------------------------------------------------
// Phase 1 (parallel): Z1[t][b][:] = x@W1[:E] + b1 (ws)   Z2 -> d_out directly
// A comes pre-converted from Xb (bf16, GEMM row order) -> pure b128 loads.
// Grid: 8 n-groups(64 cols) x 2 matmuls x 256 m-groups(8 M-tiles) = 4096 WGs
// ---------------------------------------------------------------------------
__global__ __launch_bounds__(256) void elman_xpart(
    const unsigned short* __restrict__ Xb,
    const float* __restrict__ W1, const float* __restrict__ b1,
    const float* __restrict__ W2, const float* __restrict__ b2,
    float* __restrict__ Z1, float* __restrict__ out)
{
  __shared__ unsigned sB[16384];                 // 64KB: 4 n-tiles, fragment order
  const unsigned wg    = blockIdx.x;
  const unsigned ntg   = wg & 7;                 // 64-column group
  const unsigned which = (wg >> 3) & 1;          // 0: W1 path, 1: W2 path
  const unsigned mtg   = wg >> 4;                // 0..255, 8 M-tiles each
  const float* __restrict__ W    = which ? W2 : W1;
  const float* __restrict__ bias = which ? b2 : b1;

  // Stage B = W[0:512, ntg*64 .. +63] as bf16 in WMMA fragment order.
  for (unsigned idx = threadIdx.x; idx < 16384; idx += 256) {
    unsigned nn = idx >> 12, rem = idx & 4095;
    unsigned kt = rem >> 8, L = (rem >> 3) & 31, j = rem & 7;
    unsigned k   = kt * 32 + ((j >> 2) * 16) + ((L >> 4) * 8) + 2 * (j & 3);
    unsigned col = ntg * 64 + nn * 16 + (L & 15);
    sB[idx] = pack2_bf16(W[(size_t)k * H_ + col], W[(size_t)(k + 1) * H_ + col]);
  }
  __syncthreads();

  const unsigned lane = threadIdx.x & 31;
  const unsigned wv   = threadIdx.x >> 5;        // wave -> M tile in group
  const unsigned mt   = mtg * 8 + wv;
  const unsigned arow = mt * 16 + (lane & 15);   // m = t*64 + b
  const unsigned kb   = (lane >> 4) * 8;         // A-frag K base per half-wave
  const unsigned short* __restrict__ xrow = Xb + (size_t)arow * E_;

  v8f acc[4];
#pragma unroll
  for (int i = 0; i < 4; ++i) acc[i] = (v8f){0.f,0.f,0.f,0.f,0.f,0.f,0.f,0.f};

  for (unsigned kt = 0; kt < 16; ++kt) {
    ABFrag a;
    const uint4* ap = (const uint4*)(xrow + kt * 32 + kb);
    a.q[0] = ap[0]; a.q[1] = ap[2];              // 8 bf16 @k, 8 bf16 @k+16
#pragma unroll
    for (unsigned nn = 0; nn < 4; ++nn) {
      ABFrag b;
      const uint4* bp = (const uint4*)&sB[nn * 4096 + (kt * 32 + lane) * 8];
      b.q[0] = bp[0]; b.q[1] = bp[1];
      acc[nn] = __builtin_amdgcn_wmma_f32_16x16x32_bf16(
          false, a.v, false, b.v, (short)0, acc[nn], false, false);
    }
  }

  const unsigned mloc = (lane >> 4) * 8;
#pragma unroll
  for (unsigned nn = 0; nn < 4; ++nn) {
    unsigned col = ntg * 64 + nn * 16 + (lane & 15);
    float bv = bias[col];
#pragma unroll
    for (unsigned v = 0; v < 8; ++v) {
      unsigned m = mt * 16 + mloc + v;
      unsigned b2i = m & 63, t2i = m >> 6;
      float val = acc[nn][v] + bv;
      if (which == 0)
        Z1[(size_t)t2i * (B_ * H_) + (size_t)b2i * H_ + col] = val;
      else
        out[(size_t)b2i * (T_ * O_) + (size_t)t2i * O_ + col] = val;
    }
  }
}

// ---------------------------------------------------------------------------
// Phase 2 (sequential over T): persistent 32 WGs, one 16-col slice each.
// Waves: (mt = w>>1 in 0..3 over B=64 rows, which = w&1 selects W1h/W2h).
// h kept as bf16 in a global double buffer (128KB, L2 resident).
// ---------------------------------------------------------------------------
__global__ __launch_bounds__(256) void elman_recurrent(
    const float* __restrict__ W1, const float* __restrict__ W2,
    const float* __restrict__ Z1, float* __restrict__ out,
    unsigned short* __restrict__ hbuf, unsigned* __restrict__ ctr)
{
  __shared__ unsigned sB[8192];                  // 32KB: [which][kt][lane][8 dw]
  const unsigned n0 = blockIdx.x * 16;

  // Stage hidden-part weights W[512:1024, n0..n0+15] once, bf16 fragment order.
  for (unsigned idx = threadIdx.x; idx < 8192; idx += 256) {
    unsigned wsel = idx >> 12, rem = idx & 4095;
    unsigned kt = rem >> 8, L = (rem >> 3) & 31, j = rem & 7;
    unsigned k   = 512 + kt * 32 + ((j >> 2) * 16) + ((L >> 4) * 8) + 2 * (j & 3);
    unsigned col = n0 + (L & 15);
    const float* __restrict__ W = wsel ? W2 : W1;
    sB[idx] = pack2_bf16(W[(size_t)k * H_ + col], W[(size_t)(k + 1) * H_ + col]);
  }
  __syncthreads();

  const unsigned lane  = threadIdx.x & 31;
  const unsigned wv    = threadIdx.x >> 5;
  const unsigned mt    = wv >> 1;                // M tile (batch rows)
  const unsigned which = wv & 1;
  const unsigned arow  = mt * 16 + (lane & 15);
  const unsigned kb    = (lane >> 4) * 8;
  const unsigned mloc  = (lane >> 4) * 8;
  const unsigned col   = n0 + (lane & 15);
  const unsigned sbase = which ? 4096u : 0u;

  for (unsigned t = 0; t < T_; ++t) {
    const unsigned short* __restrict__ hp = hbuf + (size_t)(t & 1) * (B_ * H_);
    unsigned short* __restrict__ hn       = hbuf + (size_t)((t + 1) & 1) * (B_ * H_);

    v8f acc = (v8f){0.f,0.f,0.f,0.f,0.f,0.f,0.f,0.f};
#pragma unroll 4
    for (unsigned kt = 0; kt < 16; ++kt) {
      ABFrag a, b;
      const uint4* ap = (const uint4*)(hp + (size_t)arow * H_ + kt * 32 + kb);
      a.q[0] = ap[0]; a.q[1] = ap[2];            // 8 bf16 @k, 8 bf16 @k+16
      const uint4* bp = (const uint4*)&sB[sbase + (kt * 32 + lane) * 8];
      b.q[0] = bp[0]; b.q[1] = bp[1];
      acc = __builtin_amdgcn_wmma_f32_16x16x32_bf16(
          false, a.v, false, b.v, (short)0, acc, false, false);
    }

    if (which == 0) {
      const float* __restrict__ zrow = Z1 + (size_t)t * (B_ * H_);
#pragma unroll
      for (unsigned v = 0; v < 8; ++v) {
        unsigned m = mt * 16 + mloc + v;
        float pre = acc[v] + zrow[(size_t)m * H_ + col];
        float hv  = 1.0f / (1.0f + __expf(-pre));
        hn[(size_t)m * H_ + col] = f32_to_bf16(hv);
        if (t == T_ - 1) out[BTO + (size_t)m * H_ + col] = hv;  // final hidden
      }
      if (t + 1 < T_)
        __builtin_prefetch(zrow + (B_ * H_) + (size_t)(mt * 16 + mloc) * H_ + col, 0, 1);
    } else {
#pragma unroll
      for (unsigned v = 0; v < 8; ++v) {
        unsigned m = mt * 16 + mloc + v;
        out[(size_t)m * (T_ * O_) + (size_t)t * O_ + col] += acc[v];  // Z2 already there
      }
    }

    // --- grid-wide barrier (monotonic counter; 32 WGs co-resident) ---
    __threadfence();
    __syncthreads();
    if (threadIdx.x == 0) {
      __hip_atomic_fetch_add(ctr, 1u, __ATOMIC_RELEASE, __HIP_MEMORY_SCOPE_AGENT);
      const unsigned target = (t + 1) * gridDim.x;
      while (__hip_atomic_load(ctr, __ATOMIC_ACQUIRE, __HIP_MEMORY_SCOPE_AGENT) < target)
        __builtin_amdgcn_s_sleep(1);
    }
    __syncthreads();
    __threadfence();
  }
}

// ---------------------------------------------------------------------------
extern "C" void kernel_launch(void* const* d_in, const int* in_sizes, int n_in,
                              void* d_out, int out_size, void* d_ws, size_t ws_size,
                              hipStream_t stream) {
  const float* x  = (const float*)d_in[0];
  const float* W1 = (const float*)d_in[1];
  const float* b1 = (const float*)d_in[2];
  const float* W2 = (const float*)d_in[3];
  const float* b2 = (const float*)d_in[4];
  float* out = (float*)d_out;

  // workspace layout:
  //   Z1   : 64MB f32  pre-activations [T][B][H]
  //   hbuf : 128KB bf16 hidden double buffer
  //   ctr  : barrier counter (padded to 1KB)
  //   Xb   : 32MB bf16 x in GEMM row order [m][k], m = t*64+b
  float*          Z1   = (float*)d_ws;
  unsigned short* hbuf = (unsigned short*)((char*)d_ws + (size_t)BTO * 4);
  unsigned*       ctr  = (unsigned*)((char*)d_ws + (size_t)BTO * 4 + 131072);
  unsigned*       Xb   = (unsigned*)((char*)d_ws + (size_t)BTO * 4 + 131072 + 1024);

  elman_init<<<64, 256, 0, stream>>>((unsigned*)hbuf, ctr);
  elman_cvt_x<<<8192, 256, 0, stream>>>(x, Xb);
  elman_xpart<<<4096, 256, 0, stream>>>((const unsigned short*)Xb, W1, b1, W2, b2, Z1, out);
  elman_recurrent<<<32, 256, 0, stream>>>(W1, W2, Z1, out, hbuf, ctr);
}
